// GraphWalkAgent_40733469835866
// MI455X (gfx1250) — matrix operand order repaired and checked by reading, not compiled, not run on hardware
//
#include <hip/hip_runtime.h>

typedef __attribute__((ext_vector_type(16))) _Float16 v16h;
typedef __attribute__((ext_vector_type(8)))  _Float16 v8h;
typedef __attribute__((ext_vector_type(8)))  float    v8f;

#define B_SZ   512
#define A_SZ   256
#define ENT_DD 256
#define REL_DD 256
#define HIST_DD 512
#define IN_DD  1024
#define ACT_DD 512
#define HUGE_V 1e31f
#define EPS_V  1e-20f

// ---------------------------------------------------------------------------
// Kernel 1: gather + concat -> X (f16)   X[b] = [E(cur[b]) | hist[b] | R(qr[b])]
// ---------------------------------------------------------------------------
__global__ __launch_bounds__(256)
void build_x_kernel(const int* __restrict__ cur, const int* __restrict__ qr,
                    const float* __restrict__ hist,
                    const float* __restrict__ ent_emb,
                    const float* __restrict__ rel_emb,
                    _Float16* __restrict__ Xh) {
  const int b = blockIdx.x;
  const int t = threadIdx.x;                 // 256 threads
  const int e = cur[b];
  const int q = qr[b];
  const float* ep = ent_emb + (size_t)e * ENT_DD;
  const float* rp = rel_emb + (size_t)q * REL_DD;
  const float* hp = hist    + (size_t)b * HIST_DD;
  _Float16* xp = Xh + (size_t)b * IN_DD;
  xp[t]        = (_Float16)ep[t];
  xp[256 + t]  = (_Float16)hp[t];
  xp[512 + t]  = (_Float16)hp[256 + t];
  xp[768 + t]  = (_Float16)rp[t];
}

// ---------------------------------------------------------------------------
// Kernel 2: weight transpose + f32->f16 convert.  in:[K,N] -> out:[N,K]
// ---------------------------------------------------------------------------
__global__ __launch_bounds__(256)
void convert_transpose_kernel(const float* __restrict__ in,
                              _Float16* __restrict__ out, int K, int N) {
  const int idx = blockIdx.x * 256 + threadIdx.x;
  if (idx >= K * N) return;
  const int k = idx / N;
  const int n = idx - k * N;
  out[(size_t)n * K + k] = (_Float16)in[idx];
}

// ---------------------------------------------------------------------------
// Kernel 3: WMMA GEMM with 1x4 register blocking.  out = act(A @ B^T + bias)
//   Am : [M,K] f16 row-major (activations)
//   Bt : [N,K] f16 row-major (transposed weights)
// One wave per 16x64 output strip: one A fragment feeds 4 independent
// v_wmma_f32_16x16x32_f16 per K-step (5 loads -> 4 WMMAs).
// A operand: 16 contiguous K per lane-half of row r, same order as B:
// lanes 0-15 hold K(k0+0..15), lanes 16-31 hold K(k0+16..31) (32B load).
// B operand: lane-half h holds 16 contiguous K values of column r (32B load).
// ---------------------------------------------------------------------------
template <bool RELU, bool OUT_F16>
__global__ __launch_bounds__(256)
void wmma_gemm_kernel(const _Float16* __restrict__ Am,
                      const _Float16* __restrict__ Bt,
                      const float*    __restrict__ bias,
                      _Float16* __restrict__ outH,
                      float*    __restrict__ outF,
                      int M, int N, int K) {
  const int wave = (blockIdx.x << 3) + (threadIdx.x >> 5);   // 8 waves / block
  const int lane = threadIdx.x & 31;
  const int ngroups = N >> 6;                // groups of 4 N-tiles (64 cols)
  const int mt = wave / ngroups;
  const int ng = wave - mt * ngroups;
  if (mt >= (M >> 4)) return;                // wave-uniform: EXEC stays full

  const int half = lane >> 4;
  const int r    = lane & 15;

  const _Float16* arow  = Am + (size_t)(mt * 16 + r) * K;
  const _Float16* brow0 = Bt + (size_t)(ng * 64 +  0 + r) * K + half * 16;
  const _Float16* brow1 = Bt + (size_t)(ng * 64 + 16 + r) * K + half * 16;
  const _Float16* brow2 = Bt + (size_t)(ng * 64 + 32 + r) * K + half * 16;
  const _Float16* brow3 = Bt + (size_t)(ng * 64 + 48 + r) * K + half * 16;

  v8f c0 = {}, c1 = {}, c2 = {}, c3 = {};
  for (int k0 = 0; k0 < K; k0 += 32) {
    v16h a = *(const v16h*)(arow + k0 + half * 16);      // 32B
    v16h b0 = *(const v16h*)(brow0 + k0);                // 32B each
    v16h b1 = *(const v16h*)(brow1 + k0);
    v16h b2 = *(const v16h*)(brow2 + k0);
    v16h b3 = *(const v16h*)(brow3 + k0);
    c0 = __builtin_amdgcn_wmma_f32_16x16x32_f16(false, a, false, b0, (short)0, c0, false, false);
    c1 = __builtin_amdgcn_wmma_f32_16x16x32_f16(false, a, false, b1, (short)0, c1, false, false);
    c2 = __builtin_amdgcn_wmma_f32_16x16x32_f16(false, a, false, b2, (short)0, c2, false, false);
    c3 = __builtin_amdgcn_wmma_f32_16x16x32_f16(false, a, false, b3, (short)0, c3, false, false);
  }

  // C/D layout: VGPR j holds M = j (lanes 0-15) / M = j+8 (lanes 16-31), N = r.
  v8f cc[4] = {c0, c1, c2, c3};
#pragma unroll
  for (int q = 0; q < 4; ++q) {
    const int col = ng * 64 + q * 16 + r;
    const float bv = bias[col];
#pragma unroll
    for (int j = 0; j < 8; ++j) {
      const int row = mt * 16 + j + half * 8;
      float v = cc[q][j] + bv;
      if (RELU) v = fmaxf(v, 0.0f);
      if (OUT_F16) outH[(size_t)row * N + col] = (_Float16)v;
      else         outF[(size_t)row * N + col] = v;
    }
  }
}

// ---------------------------------------------------------------------------
// Kernel 4: fused gather-dot scores + mask + softmax + entropy.
// One block (256 thr = 8 waves) per batch row. X2 row cached in LDS.
// Each wave computes 32 action scores with coalesced float4 row reads;
// next action's rows are prefetched (global_prefetch_b8) to hide the
// random-row gather latency that dominates this kernel.
// ---------------------------------------------------------------------------
__global__ __launch_bounds__(256)
void score_softmax_kernel(const float* __restrict__ X2,
                          const int* __restrict__ r_space,
                          const int* __restrict__ e_space,
                          const int* __restrict__ amask,
                          const float* __restrict__ ent_emb,
                          const float* __restrict__ rel_emb,
                          float* __restrict__ dist,
                          float* __restrict__ entropy) {
  const int b = blockIdx.x;
  const int t = threadIdx.x;
  const int wave = t >> 5;
  const int lane = t & 31;

  __shared__ float x2s[ACT_DD];
  __shared__ float sc[A_SZ];
  __shared__ float red[8];

  x2s[t]       = X2[(size_t)b * ACT_DD + t];
  x2s[t + 256] = X2[(size_t)b * ACT_DD + t + 256];
  __syncthreads();

  // 32 actions per wave
  for (int i = 0; i < A_SZ / 8; ++i) {
    const int a  = wave * (A_SZ / 8) + i;
    const int rr = r_space[b * A_SZ + a];
    const int er = e_space[b * A_SZ + a];
    const float4* rp = (const float4*)(rel_emb + (size_t)rr * REL_DD);
    const float4* ep = (const float4*)(ent_emb + (size_t)er * ENT_DD);

    // prefetch next action's rows (covers full cachelines, per-lane addrs)
    if (i + 1 < A_SZ / 8) {
      const int an  = a + 1;
      const int rrn = r_space[b * A_SZ + an];
      const int ern = e_space[b * A_SZ + an];
      __builtin_prefetch(rel_emb + (size_t)rrn * REL_DD + lane * 8, 0, 3);
      __builtin_prefetch(ent_emb + (size_t)ern * ENT_DD + lane * 8, 0, 3);
    }

    float s = 0.0f;
#pragma unroll
    for (int c = 0; c < REL_DD / 128; ++c) {
      const float4 v = rp[lane + 32 * c];
      const int d = (lane + 32 * c) * 4;
      s += v.x * x2s[d] + v.y * x2s[d + 1] + v.z * x2s[d + 2] + v.w * x2s[d + 3];
    }
#pragma unroll
    for (int c = 0; c < ENT_DD / 128; ++c) {
      const float4 v = ep[lane + 32 * c];
      const int d = REL_DD + (lane + 32 * c) * 4;
      s += v.x * x2s[d] + v.y * x2s[d + 1] + v.z * x2s[d + 2] + v.w * x2s[d + 3];
    }
    for (int off = 16; off > 0; off >>= 1) s += __shfl_xor(s, off, 32);
    if (lane == 0) {
      const float m = (float)amask[b * A_SZ + a];
      sc[a] = s - (1.0f - m) * HUGE_V;
    }
  }
  __syncthreads();

  // softmax over 256 scores; thread t owns action t
  const float v = sc[t];
  float wm = v;
  for (int off = 16; off > 0; off >>= 1) wm = fmaxf(wm, __shfl_xor(wm, off, 32));
  if (lane == 0) red[wave] = wm;
  __syncthreads();
  float bm = red[0];
#pragma unroll
  for (int i = 1; i < 8; ++i) bm = fmaxf(bm, red[i]);
  __syncthreads();

  const float e = __expf(v - bm);
  float wsum = e;
  for (int off = 16; off > 0; off >>= 1) wsum += __shfl_xor(wsum, off, 32);
  if (lane == 0) red[wave] = wsum;
  __syncthreads();
  float bs = 0.0f;
#pragma unroll
  for (int i = 0; i < 8; ++i) bs += red[i];
  __syncthreads();

  const float p = e / bs;
  dist[(size_t)b * A_SZ + t] = p;

  float h = -p * __logf(p + EPS_V);
  for (int off = 16; off > 0; off >>= 1) h += __shfl_xor(h, off, 32);
  if (lane == 0) red[wave] = h;
  __syncthreads();
  if (t == 0) {
    float tot = 0.0f;
#pragma unroll
    for (int i = 0; i < 8; ++i) tot += red[i];
    entropy[b] = tot;
  }
}

// ---------------------------------------------------------------------------
// Launch
// ---------------------------------------------------------------------------
extern "C" void kernel_launch(void* const* d_in, const int* in_sizes, int n_in,
                              void* d_out, int out_size, void* d_ws, size_t ws_size,
                              hipStream_t stream) {
  const int*   cur  = (const int*)d_in[0];
  const int*   qr   = (const int*)d_in[1];
  const float* hist = (const float*)d_in[2];
  const int*   rsp  = (const int*)d_in[3];
  const int*   esp  = (const int*)d_in[4];
  const int*   amsk = (const int*)d_in[5];
  const float* ee   = (const float*)d_in[6];
  const float* re   = (const float*)d_in[7];
  const float* W1   = (const float*)d_in[8];
  const float* b1   = (const float*)d_in[9];
  const float* W2   = (const float*)d_in[10];
  const float* b2   = (const float*)d_in[11];

  char* ws = (char*)d_ws;
  _Float16* Xh  = (_Float16*)(ws + 0);         // 512*1024*2 = 1.0 MB
  _Float16* W1T = (_Float16*)(ws + 1048576);   // 512*1024*2 = 1.0 MB
  _Float16* W2T = (_Float16*)(ws + 2097152);   // 512*512*2  = 0.5 MB
  _Float16* H   = (_Float16*)(ws + 2621440);   // 512*512*2  = 0.5 MB
  float*    X2  = (float*)   (ws + 3145728);   // 512*512*4  = 1.0 MB

  // 1) build f16 input activations
  build_x_kernel<<<B_SZ, 256, 0, stream>>>(cur, qr, hist, ee, re, Xh);

  // 2) convert+transpose weights (tiny: 3 MB total)
  convert_transpose_kernel<<<(IN_DD * ACT_DD + 255) / 256, 256, 0, stream>>>(
      W1, W1T, IN_DD, ACT_DD);
  convert_transpose_kernel<<<(ACT_DD * ACT_DD + 255) / 256, 256, 0, stream>>>(
      W2, W2T, ACT_DD, ACT_DD);

  // 3) H = relu(X @ W1 + b1)   [512x512], WMMA f16->f32, 1x4 blocked
  const int jobs = (B_SZ / 16) * (ACT_DD / 64);   // 32 * 8 = 256 wave-jobs
  wmma_gemm_kernel<true, true><<<jobs / 8, 256, 0, stream>>>(
      Xh, W1T, b1, H, nullptr, B_SZ, ACT_DD, IN_DD);

  // 4) X2 = H @ W2 + b2   [512x512] f32 out
  wmma_gemm_kernel<false, false><<<jobs / 8, 256, 0, stream>>>(
      H, W2T, b2, nullptr, X2, B_SZ, ACT_DD, ACT_DD);

  // 5) fused gather-dot scores + mask + softmax + entropy
  float* out = (float*)d_out;
  score_softmax_kernel<<<B_SZ, 256, 0, stream>>>(
      X2, rsp, esp, amsk, ee, re, out, out + (size_t)B_SZ * A_SZ);
}
